// MultiHeadAttention_21088289423530
// MI455X (gfx1250) — compile-verified, matrix operand order checked by reading, not run on hardware
//
#include <hip/hip_runtime.h>
#include <hip/hip_bf16.h>

typedef __attribute__((ext_vector_type(16))) __bf16 v16bf;
typedef __attribute__((ext_vector_type(8)))  __bf16 v8bf;
typedef __attribute__((ext_vector_type(8)))  float  v8f;

#define HW    4096
#define NH    8
#define HD    32
#define CDIM  256
#define BATCH 2

// Per-lane K-index pattern for 16-bit A/B WMMA fragments (16x32 / 32x16):
// lanes 0-15: VGPR0-3 -> K=0..7, VGPR4-7 -> K=16..23 ; lanes 16-31: +8.
// => per lane, fragment elements are two contiguous 16-byte runs:
//    elements 0..7  at k = 8*half      (16B)
//    elements 8..15 at k = 16 + 8*half (16B)
__device__ __forceinline__ int kmap(int e, int half) {
  return (e < 8 ? e : e + 8) + (half << 3);
}

__device__ __forceinline__ __bf16 to_bf16(float f) { return (__bf16)f; }

union Frag16 { v16bf v; v8bf h[2]; };

// ---------------------------------------------------------------------------
// 1x1-conv projection GEMM: Y[o,p] = sum_c W[o,c] * X[c,p] + bias[o], scaled.
// MODE 0: bf16 out, layout [b, h, p, d]   (Q and K)
// MODE 1: bf16 out, layout [b, h, d, p]   (V, transposed)
// MODE 2: fp32 out, layout [b, o, p]      (final output projection)
// ---------------------------------------------------------------------------
template <typename Tin, int MODE>
__global__ void proj_kernel(const Tin* __restrict__ x, const float* __restrict__ w,
                            const float* __restrict__ bias, void* __restrict__ outp,
                            float scale) {
  const int lane = threadIdx.x & 31;
  const int half = lane >> 4;
  const int mn   = lane & 15;

  int idx = blockIdx.x;
  const int b = idx / ((CDIM / 16) * (HW / 16));
  idx %= (CDIM / 16) * (HW / 16);
  const int o0 = (idx / (HW / 16)) * 16;
  const int p0 = (idx % (HW / 16)) * 16;

  const Tin* xb = x + (size_t)b * CDIM * HW;

  v8f acc = {};
#pragma unroll
  for (int c0 = 0; c0 < CDIM; c0 += 32) {
    v16bf afrag, bfrag;
#pragma unroll
    for (int e = 0; e < 16; ++e) {
      const int kk = kmap(e, half);
      afrag[e] = to_bf16(w[(size_t)(o0 + mn) * CDIM + c0 + kk]);
      bfrag[e] = to_bf16((float)xb[(size_t)(c0 + kk) * HW + p0 + mn]);
    }
    acc = __builtin_amdgcn_wmma_f32_16x16x32_bf16(false, afrag, false, bfrag,
                                                  (short)0, acc, false, false);
  }

#pragma unroll
  for (int r = 0; r < 8; ++r) {
    const int o = o0 + r + (half << 3);   // output channel
    const int p = p0 + mn;                // spatial position
    const float val = (acc[r] + bias[o]) * scale;
    if (MODE == 0) {
      __bf16* q = (__bf16*)outp;
      const int h = o >> 5, d = o & 31;
      q[(((size_t)b * NH + h) * HW + p) * HD + d] = to_bf16(val);
    } else if (MODE == 1) {
      __bf16* vt = (__bf16*)outp;
      const int h = o >> 5, d = o & 31;
      vt[(((size_t)b * NH + h) * HD + d) * HW + p] = to_bf16(val);
    } else {
      float* y = (float*)outp;
      y[((size_t)b * CDIM + o) * HW + p] = val;
    }
  }
}

// ---------------------------------------------------------------------------
// Flash attention with cooperative LDS tiling.
// Block = 128 threads = 4 waves; each wave owns 32 query columns (2 i-sets),
// the block covers 128. Per 64-row j-chunk:
//   - K chunk (64x32 bf16, 4KB, contiguous) and V chunk (32x64 bf16, 4KB)
//     staged into LDS ONCE PER BLOCK (global traffic / 4 vs per-wave loads),
//     software-pipelined: next chunk's global loads issue between barriers.
//   - fragments read from LDS as two ds_load_b128 per fragment.
//   - per wave: 8 score WMMAs (4 K-tiles x 2 i-sets), online softmax with
//     in-register P-fragment assembly, 8 PV WMMAs.
// ---------------------------------------------------------------------------
__global__ void __launch_bounds__(128, 1)
flash_attn_kernel(const __bf16* __restrict__ qbuf,
                  const __bf16* __restrict__ kbuf,
                  const __bf16* __restrict__ vbuf,
                  __bf16* __restrict__ att) {
  __shared__ __bf16 Kl[64 * 32];     // [j-row][d]
  __shared__ __bf16 Vl[32 * 64];     // [d][j-col]

  const int tid  = threadIdx.x;      // 0..127
  const int wave = tid >> 5;         // 0..3
  const int lane = tid & 31;
  const int half = lane >> 4;
  const int mn   = lane & 15;

  const int bh = blockIdx.x / (HW / 128);
  const int i0 = (blockIdx.x % (HW / 128)) * 128 + wave * 32;
  const int b  = bh / NH;
  const int h  = bh % NH;

  const __bf16* qb = qbuf + (size_t)bh * HW * HD;
  const __bf16* kb = kbuf + (size_t)bh * HW * HD;
  const __bf16* vb = vbuf + (size_t)bh * HD * HW;

  // staging assignment for the cooperative V copy: 32 rows x 8 16B-segments
  const int vrow = tid & 31;
  const int vseg = tid >> 5;

  // Q as B-operand fragments: two 16B runs per lane, straight from global.
  Frag16 qf0, qf1;
  qf0.h[0] = *(const v8bf*)(qb + (size_t)(i0 + mn) * HD + 8 * half);
  qf0.h[1] = *(const v8bf*)(qb + (size_t)(i0 + mn) * HD + 16 + 8 * half);
  qf1.h[0] = *(const v8bf*)(qb + (size_t)(i0 + 16 + mn) * HD + 8 * half);
  qf1.h[1] = *(const v8bf*)(qb + (size_t)(i0 + 16 + mn) * HD + 16 + 8 * half);

  v8f ot00 = {}, ot01 = {};        // i-set 0: d = 0..15 / 16..31
  v8f ot10 = {}, ot11 = {};        // i-set 1: d = 0..15 / 16..31
  float m0 = -1e30f, l0 = 0.0f;
  float m1 = -1e30f, l1 = 0.0f;

  // prologue: stage chunk j0 = 0 into registers
  v8bf kst0 = *(const v8bf*)(kb + tid * 8);
  v8bf kst1 = *(const v8bf*)(kb + 1024 + tid * 8);
  v8bf vst0 = *(const v8bf*)(vb + (size_t)vrow * HW + vseg * 8);
  v8bf vst1 = *(const v8bf*)(vb + (size_t)vrow * HW + (vseg + 4) * 8);

  for (int j0 = 0; j0 < HW; j0 += 64) {
    __syncthreads();               // previous iteration's LDS reads are done
    *(v8bf*)&Kl[tid * 8]                    = kst0;
    *(v8bf*)&Kl[1024 + tid * 8]             = kst1;
    *(v8bf*)&Vl[vrow * 64 + vseg * 8]       = vst0;
    *(v8bf*)&Vl[vrow * 64 + (vseg + 4) * 8] = vst1;

    // issue next chunk's global loads (wrap keeps addresses valid + uniform)
    const int jn = (j0 + 64) & (HW - 1);
    kst0 = *(const v8bf*)(kb + (size_t)jn * HD + tid * 8);
    kst1 = *(const v8bf*)(kb + (size_t)jn * HD + 1024 + tid * 8);
    vst0 = *(const v8bf*)(vb + (size_t)vrow * HW + jn + vseg * 8);
    vst1 = *(const v8bf*)(vb + (size_t)vrow * HW + jn + (vseg + 4) * 8);
    __syncthreads();               // LDS chunk ready

    // ---- scores: each K fragment (2 ds_load_b128) feeds both i-sets ----
    v8f s0[4], s1[4];
#pragma unroll
    for (int t = 0; t < 4; ++t) {
      Frag16 kf;
      const int row = t * 16 + mn;
      kf.h[0] = *(const v8bf*)&Kl[row * 32 + 8 * half];
      kf.h[1] = *(const v8bf*)&Kl[row * 32 + 16 + 8 * half];
      v8f z0 = {}, z1 = {};
      s0[t] = __builtin_amdgcn_wmma_f32_16x16x32_bf16(false, kf.v, false, qf0.v,
                                                      (short)0, z0, false, false);
      s1[t] = __builtin_amdgcn_wmma_f32_16x16x32_bf16(false, kf.v, false, qf1.v,
                                                      (short)0, z1, false, false);
    }

    v16bf p01_0, p23_0, p01_1, p23_1;

    // ---- online softmax + P fragments, i-set 0 ----
    {
      float cm = s0[0][0];
#pragma unroll
      for (int t = 0; t < 4; ++t)
#pragma unroll
        for (int r = 0; r < 8; ++r) cm = fmaxf(cm, s0[t][r]);
      cm = fmaxf(cm, __shfl_xor(cm, 16, 32));
      const float nm   = fmaxf(m0, cm);
      const float corr = __expf(m0 - nm);
      float psum = 0.0f;
#pragma unroll
      for (int t = 0; t < 4; ++t)
#pragma unroll
        for (int r = 0; r < 8; ++r) {
          const float p = __expf(s0[t][r] - nm);
          s0[t][r] = p;
          psum += p;
        }
      psum += __shfl_xor(psum, 16, 32);
      l0 = l0 * corr + psum;
      m0 = nm;
#pragma unroll
      for (int r = 0; r < 8; ++r) { ot00[r] *= corr; ot01[r] *= corr; }
#pragma unroll
      for (int e = 0; e < 8; ++e) {
        p01_0[e]     = to_bf16(s0[0][e]);
        p01_0[e + 8] = to_bf16(s0[1][e]);
        p23_0[e]     = to_bf16(s0[2][e]);
        p23_0[e + 8] = to_bf16(s0[3][e]);
      }
    }
    // ---- online softmax + P fragments, i-set 1 ----
    {
      float cm = s1[0][0];
#pragma unroll
      for (int t = 0; t < 4; ++t)
#pragma unroll
        for (int r = 0; r < 8; ++r) cm = fmaxf(cm, s1[t][r]);
      cm = fmaxf(cm, __shfl_xor(cm, 16, 32));
      const float nm   = fmaxf(m1, cm);
      const float corr = __expf(m1 - nm);
      float psum = 0.0f;
#pragma unroll
      for (int t = 0; t < 4; ++t)
#pragma unroll
        for (int r = 0; r < 8; ++r) {
          const float p = __expf(s1[t][r] - nm);
          s1[t][r] = p;
          psum += p;
        }
      psum += __shfl_xor(psum, 16, 32);
      l1 = l1 * corr + psum;
      m1 = nm;
#pragma unroll
      for (int r = 0; r < 8; ++r) { ot10[r] *= corr; ot11[r] *= corr; }
#pragma unroll
      for (int e = 0; e < 8; ++e) {
        p01_1[e]     = to_bf16(s1[0][e]);
        p01_1[e + 8] = to_bf16(s1[1][e]);
        p23_1[e]     = to_bf16(s1[2][e]);
        p23_1[e + 8] = to_bf16(s1[3][e]);
      }
    }

    // ---- O^T += V_tile x P : each V fragment (2 ds_load_b128) feeds both ----
#pragma unroll
    for (int dt = 0; dt < 2; ++dt) {       // d-tile: 0..15 / 16..31
#pragma unroll
      for (int jh = 0; jh < 2; ++jh) {     // j half: k = 0..31 / 32..63
        Frag16 vf;
        const int row = dt * 16 + mn;
        vf.h[0] = *(const v8bf*)&Vl[row * 64 + jh * 32 + 8 * half];
        vf.h[1] = *(const v8bf*)&Vl[row * 64 + jh * 32 + 16 + 8 * half];
        const v16bf pA = jh ? p23_0 : p01_0;
        const v16bf pB = jh ? p23_1 : p01_1;
        if (dt == 0) {
          ot00 = __builtin_amdgcn_wmma_f32_16x16x32_bf16(false, vf.v, false, pA,
                                                         (short)0, ot00, false, false);
          ot10 = __builtin_amdgcn_wmma_f32_16x16x32_bf16(false, vf.v, false, pB,
                                                         (short)0, ot10, false, false);
        } else {
          ot01 = __builtin_amdgcn_wmma_f32_16x16x32_bf16(false, vf.v, false, pA,
                                                         (short)0, ot01, false, false);
          ot11 = __builtin_amdgcn_wmma_f32_16x16x32_bf16(false, vf.v, false, pB,
                                                         (short)0, ot11, false, false);
        }
      }
    }
  }

  // ---- normalize and store O^T: att[b, c = h*32+d, p = i] in bf16 ----
  const size_t cbase = (size_t)b * CDIM + h * HD;
  const float  inv0  = 1.0f / l0;
  const float  inv1  = 1.0f / l1;
#pragma unroll
  for (int r = 0; r < 8; ++r) {
    const int d = r + (half << 3);
    att[(cbase + d) * HW + i0 + mn]           = to_bf16(ot00[r] * inv0);
    att[(cbase + 16 + d) * HW + i0 + mn]      = to_bf16(ot01[r] * inv0);
    att[(cbase + d) * HW + i0 + 16 + mn]      = to_bf16(ot10[r] * inv1);
    att[(cbase + 16 + d) * HW + i0 + 16 + mn] = to_bf16(ot11[r] * inv1);
  }
}

// ---------------------------------------------------------------------------
extern "C" void kernel_launch(void* const* d_in, const int* in_sizes, int n_in,
                              void* d_out, int out_size, void* d_ws, size_t ws_size,
                              hipStream_t stream) {
  const float* query = (const float*)d_in[0];
  const float* key_  = (const float*)d_in[1];
  const float* value = (const float*)d_in[2];
  const float* wq = (const float*)d_in[3];
  const float* bq = (const float*)d_in[4];
  const float* wk = (const float*)d_in[5];
  const float* bk = (const float*)d_in[6];
  const float* wv = (const float*)d_in[7];
  const float* bv = (const float*)d_in[8];
  const float* wo = (const float*)d_in[9];
  const float* bo = (const float*)d_in[10];

  const size_t QKV = (size_t)BATCH * NH * HW * HD;  // 2,097,152 bf16 elems each
  __bf16* qbuf = (__bf16*)d_ws;
  __bf16* kbuf = qbuf + QKV;
  __bf16* vbuf = kbuf + QKV;
  __bf16* attb = vbuf + QKV;                        // [b, c, p] bf16

  const int proj_blocks  = BATCH * (CDIM / 16) * (HW / 16);  // 8192 waves
  const int flash_blocks = BATCH * NH * (HW / 128);          // 512 blocks x 4 waves
  const float qscale = 0.17677669529663687f;                 // 1/sqrt(HD)

  proj_kernel<float, 0><<<proj_blocks, 32, 0, stream>>>(query, wq, bq, (void*)qbuf, qscale);
  proj_kernel<float, 0><<<proj_blocks, 32, 0, stream>>>(key_,  wk, bk, (void*)kbuf, 1.0f);
  proj_kernel<float, 1><<<proj_blocks, 32, 0, stream>>>(value, wv, bv, (void*)vbuf, 1.0f);

  flash_attn_kernel<<<flash_blocks, 128, 0, stream>>>(qbuf, kbuf, vbuf, attb);

  proj_kernel<__bf16, 2><<<proj_blocks, 32, 0, stream>>>(attb, wo, bo, d_out, 1.0f);
}
